// GaussianGraphSAGE_59184649339057
// MI455X (gfx1250) — matrix operand order repaired and checked by reading, not compiled, not run on hardware
//
#include <hip/hip_runtime.h>
#include <hip/hip_bf16.h>
#include <math.h>

// ---------------------------------------------------------------------------
// GaussianGraphSAGE for MI455X (gfx1250): bf16 activations + v_wmma_f32_16x16x32_bf16
// LDS-staged weight operands, templated/unrolled K-loop, prefetched A-stream.
// ---------------------------------------------------------------------------

#define NN      100000
#define EE      1600000
#define GG      512
#define IN_CH   128
#define HH      64
#define FCD     128
#define CC      2

typedef __attribute__((ext_vector_type(16))) __bf16 bf16x16;
typedef __attribute__((ext_vector_type(8)))  __bf16 bf16x8;
typedef __attribute__((ext_vector_type(4)))  __bf16 bf16x4;
typedef __attribute__((ext_vector_type(8)))  float  f32x8;

// ---------------------------------------------------------------------------
// f32 -> bf16 convert
__global__ void k_f2bf(const float* __restrict__ in, __bf16* __restrict__ out, long n) {
    long i = (long)blockIdx.x * blockDim.x + threadIdx.x;
    if (i < n) out[i] = (__bf16)in[i];
}

// ---------------------------------------------------------------------------
// Pack a [K,64] f32 weight matrix into WMMA B-operand layout (bf16).
// Block layout: [kt][nt][lane] each 16 bf16 contiguous.
// lane 0-15 : col = nt*16 + lane,    K = kt*32 + 0..7, 16..23
// lane 16-31: col = nt*16 + lane-16, K = kt*32 + 8..15, 24..31
__global__ void k_pack_w(const float* __restrict__ W, __bf16* __restrict__ out, int K) {
    int t = blockIdx.x * blockDim.x + threadIdx.x;
    int nK = K >> 5;
    int total = nK * 4 * 32;
    if (t >= total) return;
    int lane = t & 31;
    int nt   = (t >> 5) & 3;
    int kt   = t >> 7;
    int col  = (lane & 15) + nt * 16;
    int kb   = kt * 32 + (lane >> 4) * 8;
    __bf16* o = out + (long)t * 16;
#pragma unroll
    for (int j = 0; j < 16; ++j)
        o[j] = (__bf16)W[(long)(kb + (j < 8 ? j : j + 8)) * HH + col];
}

// ---------------------------------------------------------------------------
// Edge degree count (once; same graph for all layers)
__global__ void k_edge_count(const int* __restrict__ dst, float* __restrict__ cnt, int E) {
    long e = (long)blockIdx.x * blockDim.x + threadIdx.x;
    if (e < E) atomicAdd(&cnt[dst[e]], 1.0f);
}

// ---------------------------------------------------------------------------
// Scatter-add neighbor features: agg[dst] += x[src]  (bf16 gather, f32 atomics)
// One thread handles 4 channels of one edge.
__global__ void k_edge_agg(const __bf16* __restrict__ x, const int* __restrict__ src,
                           const int* __restrict__ dst, float* __restrict__ agg,
                           int E, int d) {
    long t = (long)blockIdx.x * blockDim.x + threadIdx.x;
    int perEdge = d >> 2;
    long e = t / perEdge;
    if (e >= E) return;
    int c = (int)(t - e * perEdge) * 4;
    int s = src[e], dn = dst[e];
    bf16x4 xv = *(const bf16x4*)(x + (long)s * d + c);
    float* ad = agg + (long)dn * d + c;
    atomicAdd(ad + 0, (float)xv[0]);
    atomicAdd(ad + 1, (float)xv[1]);
    atomicAdd(ad + 2, (float)xv[2]);
    atomicAdd(ad + 3, (float)xv[3]);
}

// ---------------------------------------------------------------------------
// agg /= max(cnt,1)  and convert to bf16
__global__ void k_normalize(const float* __restrict__ agg, const float* __restrict__ cnt,
                            __bf16* __restrict__ aggb, int n, int d) {
    long t = (long)blockIdx.x * blockDim.x + threadIdx.x;
    long total = (long)n * d;
    if (t >= total) return;
    long i = t / d;
    float c = cnt[i];
    aggb[t] = (__bf16)(agg[t] / fmaxf(c, 1.0f));
}

// ---------------------------------------------------------------------------
// WMMA GEMM: out[row, 0:64] = relu( agg[row,:]@Wl + x[row,:]@Wr + bias )
// One wave computes a 16x64 tile: 4 f32 accumulators, 2 WMMAs per (kt,nt).
// Weights (B operands) staged in LDS once per workgroup; A streamed from global.
// A fragment (16-bit, 16x32):
//   lane 0-15 : row = lane,    h[0]=K kt*32+0..7,  h[1]=K kt*32+16..23
//   lane 16-31: row = lane-16, h[0]=K kt*32+8..15, h[1]=K kt*32+24..31
template <int DK>
__global__ __launch_bounds__(256)
void k_sage_gemm(const __bf16* __restrict__ agg, const __bf16* __restrict__ xin,
                 const __bf16* __restrict__ WlP, const __bf16* __restrict__ WrP,
                 const float* __restrict__ bias, __bf16* __restrict__ actOut,
                 float* __restrict__ f32Out, int nRows) {
    constexpr int NK = DK / 32;            // k-tiles
    constexpr int NVEC = NK * 4 * 32;      // bf16x16 vectors per weight matrix
    __shared__ bf16x16 sBl[NVEC];
    __shared__ bf16x16 sBr[NVEC];

    // cooperative stage of both packed weight matrices into LDS
    for (int i = threadIdx.x; i < NVEC; i += 256) {
        sBl[i] = ((const bf16x16*)WlP)[i];
        sBr[i] = ((const bf16x16*)WrP)[i];
    }
    __syncthreads();

    const int lane = threadIdx.x & 31;
    const int wave = threadIdx.x >> 5;
    const long rowBase = ((long)blockIdx.x * 8 + wave) * 16;
    const bool valid = rowBase < nRows;

    // clamp so tail waves load safely and still hit the barrier above
    long aRowIdx = rowBase + (lane & 15);
    if (aRowIdx >= nRows) aRowIdx = nRows - 1;

    f32x8 acc[4] = {};
    const int kA = (lane >> 4) * 8;
    const __bf16* aPtr = agg + aRowIdx * DK + kA;
    const __bf16* xPtr = xin + aRowIdx * DK + kA;

    // prefetch the next row-tile's A-stream into cache
    __builtin_prefetch(aPtr + 16L * DK, 0, 1);
    __builtin_prefetch(xPtr + 16L * DK, 0, 1);

    union AFrag { bf16x16 v; bf16x8 h[2]; };

#pragma unroll
    for (int kt = 0; kt < NK; ++kt) {
        AFrag a, xa;
        a.h[0]  = *(const bf16x8*)(aPtr + kt * 32);
        a.h[1]  = *(const bf16x8*)(aPtr + kt * 32 + 16);
        xa.h[0] = *(const bf16x8*)(xPtr + kt * 32);
        xa.h[1] = *(const bf16x8*)(xPtr + kt * 32 + 16);
#pragma unroll
        for (int nt = 0; nt < 4; ++nt) {
            bf16x16 bL = sBl[(kt * 4 + nt) * 32 + lane];
            bf16x16 bR = sBr[(kt * 4 + nt) * 32 + lane];
            acc[nt] = __builtin_amdgcn_wmma_f32_16x16x32_bf16(
                false, a.v, false, bL, (short)0, acc[nt], false, false);
            acc[nt] = __builtin_amdgcn_wmma_f32_16x16x32_bf16(
                false, xa.v, false, bR, (short)0, acc[nt], false, false);
        }
    }

    if (!valid) return;

    // C/D layout: lane 0-15 -> col=lane, rows v+0..7 in VGPR v; lane 16-31 -> rows v+8
    const int col0 = lane & 15;
    const int rOff = (lane >> 4) * 8;
#pragma unroll
    for (int nt = 0; nt < 4; ++nt) {
        const int col = nt * 16 + col0;
        const float b = bias[col];
#pragma unroll
        for (int v = 0; v < 8; ++v) {
            float val = acc[nt][v] + b;
            val = fmaxf(val, 0.0f);
            long idx = (rowBase + rOff + v) * HH + col;
            actOut[idx] = (__bf16)val;
            if (f32Out) f32Out[idx] = val;
        }
    }
}

// ---------------------------------------------------------------------------
// z = mean + eps * exp(0.5*logvar); segment-sum into gsum/gcnt by batch id
__global__ void k_reparam_pool(const float* __restrict__ meanF, const float* __restrict__ lvF,
                               const float* __restrict__ eps, const int* __restrict__ batch,
                               float* __restrict__ gsum, float* __restrict__ gcnt, int n) {
    long t = (long)blockIdx.x * blockDim.x + threadIdx.x;
    if (t >= (long)n * HH) return;
    long i = t >> 6;
    int  c = (int)(t & 63);
    float z = meanF[t] + eps[t] * expf(0.5f * lvF[t]);
    int g = batch[i];
    atomicAdd(&gsum[(long)g * HH + c], z);
    if (c == 0) atomicAdd(&gcnt[g], 1.0f);
}

// ---------------------------------------------------------------------------
// pooled -> fc1(relu) -> fc2 -> log_softmax.  One block per graph.
__global__ void k_head(const float* __restrict__ gsum, const float* __restrict__ gcnt,
                       const float* __restrict__ fc1W, const float* __restrict__ fc1b,
                       const float* __restrict__ fc2W, const float* __restrict__ fc2b,
                       float* __restrict__ out) {
    __shared__ float pooled[HH];
    __shared__ float h[FCD];
    const int g = blockIdx.x;
    const float inv = 1.0f / fmaxf(gcnt[g], 1.0f);
    if (threadIdx.x < HH) pooled[threadIdx.x] = gsum[(long)g * HH + threadIdx.x] * inv;
    __syncthreads();
    const int f = threadIdx.x;
    float s = fc1b[f];
    for (int k = 0; k < HH; ++k) s += pooled[k] * fc1W[k * FCD + f];
    h[f] = fmaxf(s, 0.0f);
    __syncthreads();
    if (threadIdx.x == 0) {
        float l0 = fc2b[0], l1 = fc2b[1];
        for (int k = 0; k < FCD; ++k) { l0 += h[k] * fc2W[k * CC + 0]; l1 += h[k] * fc2W[k * CC + 1]; }
        float m = fmaxf(l0, l1);
        float lse = m + logf(expf(l0 - m) + expf(l1 - m));
        out[(long)g * CC + 0] = l0 - lse;
        out[(long)g * CC + 1] = l1 - lse;
    }
}

// ---------------------------------------------------------------------------
static inline size_t alignUp(size_t v) { return (v + 255) & ~(size_t)255; }

extern "C" void kernel_launch(void* const* d_in, const int* in_sizes, int n_in,
                              void* d_out, int out_size, void* d_ws, size_t ws_size,
                              hipStream_t stream) {
    const float* x        = (const float*)d_in[0];
    const float* eps      = (const float*)d_in[1];
    const float* m0_Wl    = (const float*)d_in[2];
    const float* m0_bl    = (const float*)d_in[3];
    const float* m0_Wr    = (const float*)d_in[4];
    const float* mR_Wl    = (const float*)d_in[5];
    const float* mR_bl    = (const float*)d_in[6];
    const float* mR_Wr    = (const float*)d_in[7];
    const float* v0_Wl    = (const float*)d_in[8];
    const float* v0_bl    = (const float*)d_in[9];
    const float* v0_Wr    = (const float*)d_in[10];
    const float* vR_Wl    = (const float*)d_in[11];
    const float* vR_bl    = (const float*)d_in[12];
    const float* vR_Wr    = (const float*)d_in[13];
    const float* fc1_W    = (const float*)d_in[14];
    const float* fc1_b    = (const float*)d_in[15];
    const float* fc2_W    = (const float*)d_in[16];
    const float* fc2_b    = (const float*)d_in[17];
    const int*   eidx     = (const int*)d_in[18];
    const int*   batch    = (const int*)d_in[19];
    const int*   esrc = eidx;
    const int*   edst = eidx + EE;

    float* outLogits = (float*)d_out;                       // [G,2]
    float* outMean   = outLogits + (size_t)GG * CC;         // [N,64]
    float* outLvar   = outMean + (size_t)NN * HH;           // [N,64]

    // ---- workspace carve-out ----
    char* ws = (char*)d_ws;
    size_t cur = 0;
    auto take = [&](size_t bytes) { char* p = ws + cur; cur += alignUp(bytes); return p; };
    __bf16* xb    = (__bf16*)take((size_t)NN * IN_CH * 2);   // x in bf16
    float*  agg   = (float*) take((size_t)NN * IN_CH * 4);   // f32 neighbor sums
    float*  cnt   = (float*) take((size_t)NN * 4);           // degrees
    __bf16* aggb  = (__bf16*)take((size_t)NN * IN_CH * 2);   // normalized bf16 agg
    __bf16* actM  = (__bf16*)take((size_t)NN * HH * 2);      // mean-branch activations
    __bf16* actV  = (__bf16*)take((size_t)NN * HH * 2);      // var-branch activations
    __bf16* pw    = (__bf16*)take((size_t)65536 * 2);        // packed weights
    float*  gsum  = (float*) take((size_t)GG * HH * 4);
    float*  gcnt  = (float*) take((size_t)GG * 4);
    (void)ws_size; (void)in_sizes; (void)n_in; (void)out_size;

    // packed-weight sub-offsets (elements)
    const size_t PW128 = 8192, PW64 = 4096;
    __bf16* pw_m0l = pw;                 __bf16* pw_m0r = pw_m0l + PW128;
    __bf16* pw_v0l = pw_m0r + PW128;     __bf16* pw_v0r = pw_v0l + PW128;
    __bf16* pw_mRl[2] = { pw_v0r + PW128,          pw_v0r + PW128 + PW64 };
    __bf16* pw_mRr[2] = { pw_mRl[1] + PW64,        pw_mRl[1] + 2 * PW64 };
    __bf16* pw_vRl[2] = { pw_mRr[1] + PW64,        pw_mRr[1] + 2 * PW64 };
    __bf16* pw_vRr[2] = { pw_vRl[1] + PW64,        pw_vRl[1] + 2 * PW64 };

    const int T = 256;

    // 1) x -> bf16
    {
        long n = (long)NN * IN_CH;
        k_f2bf<<<(unsigned)((n + T - 1) / T), T, 0, stream>>>(x, xb, n);
    }

    // 2) pack all weights into WMMA B layout
    k_pack_w<<<(4 * 4 * 32 + T - 1) / T, T, 0, stream>>>(m0_Wl, pw_m0l, IN_CH);
    k_pack_w<<<(4 * 4 * 32 + T - 1) / T, T, 0, stream>>>(m0_Wr, pw_m0r, IN_CH);
    k_pack_w<<<(4 * 4 * 32 + T - 1) / T, T, 0, stream>>>(v0_Wl, pw_v0l, IN_CH);
    k_pack_w<<<(4 * 4 * 32 + T - 1) / T, T, 0, stream>>>(v0_Wr, pw_v0r, IN_CH);
    for (int i = 0; i < 2; ++i) {
        k_pack_w<<<(2 * 4 * 32 + T - 1) / T, T, 0, stream>>>(mR_Wl + (size_t)i * HH * HH, pw_mRl[i], HH);
        k_pack_w<<<(2 * 4 * 32 + T - 1) / T, T, 0, stream>>>(mR_Wr + (size_t)i * HH * HH, pw_mRr[i], HH);
        k_pack_w<<<(2 * 4 * 32 + T - 1) / T, T, 0, stream>>>(vR_Wl + (size_t)i * HH * HH, pw_vRl[i], HH);
        k_pack_w<<<(2 * 4 * 32 + T - 1) / T, T, 0, stream>>>(vR_Wr + (size_t)i * HH * HH, pw_vRr[i], HH);
    }

    // 3) degree counts (reused by every layer)
    hipMemsetAsync(cnt, 0, (size_t)NN * 4, stream);
    k_edge_count<<<(EE + T - 1) / T, T, 0, stream>>>(edst, cnt, EE);

    const int gemmBlocks = (int)(((long)NN / 16 + 7) / 8);   // 8 waves/block, 16 rows/wave

    // helper: one SAGE layer
    auto sageLayer = [&](const __bf16* xin, int dK, const __bf16* Wl, const __bf16* Wr,
                         const float* bl, __bf16* actOut, float* f32Out) {
        hipMemsetAsync(agg, 0, (size_t)NN * dK * 4, stream);
        long nt = (long)EE * (dK >> 2);
        k_edge_agg<<<(unsigned)((nt + T - 1) / T), T, 0, stream>>>(xin, esrc, edst, agg, EE, dK);
        long nn = (long)NN * dK;
        k_normalize<<<(unsigned)((nn + T - 1) / T), T, 0, stream>>>(agg, cnt, aggb, NN, dK);
        if (dK == IN_CH)
            k_sage_gemm<IN_CH><<<gemmBlocks, 256, 0, stream>>>(aggb, xin, Wl, Wr, bl, actOut, f32Out, NN);
        else
            k_sage_gemm<HH><<<gemmBlocks, 256, 0, stream>>>(aggb, xin, Wl, Wr, bl, actOut, f32Out, NN);
    };

    // 4) mean branch
    sageLayer(xb, IN_CH, pw_m0l, pw_m0r, m0_bl, actM, nullptr);
    sageLayer(actM, HH, pw_mRl[0], pw_mRr[0], mR_bl + 0 * HH, actM, nullptr);
    sageLayer(actM, HH, pw_mRl[1], pw_mRr[1], mR_bl + 1 * HH, actM, outMean);

    // 5) var branch
    sageLayer(xb, IN_CH, pw_v0l, pw_v0r, v0_bl, actV, nullptr);
    sageLayer(actV, HH, pw_vRl[0], pw_vRr[0], vR_bl + 0 * HH, actV, nullptr);
    sageLayer(actV, HH, pw_vRl[1], pw_vRr[1], vR_bl + 1 * HH, actV, outLvar);

    // 6) reparameterize + global mean pool
    hipMemsetAsync(gsum, 0, (size_t)GG * HH * 4, stream);
    hipMemsetAsync(gcnt, 0, (size_t)GG * 4, stream);
    {
        long n = (long)NN * HH;
        k_reparam_pool<<<(unsigned)((n + T - 1) / T), T, 0, stream>>>(
            outMean, outLvar, eps, batch, gsum, gcnt, NN);
    }

    // 7) MLP head + log_softmax
    k_head<<<GG, FCD, 0, stream>>>(gsum, gcnt, fc1_W, fc1_b, fc2_W, fc2_b, outLogits);
}
